// MoE_58548994179550
// MI455X (gfx1250) — matrix operand order repaired and checked by reading, not compiled, not run on hardware
//
#include <hip/hip_runtime.h>

#define DEVINL __device__ __forceinline__

constexpr int Btok = 16384;
constexpr int Dim  = 1024;
constexpr int Hdim = 4096;
constexpr int Odim = 1024;
constexpr int Eexp = 8;
constexpr int Ktop = 2;
constexpr int Cap  = 5120;
constexpr int Nrt  = Btok * Ktop;   // 32768 routed slots
constexpr int NBLK = Nrt / 1024;    // 32 routing blocks
constexpr int BK   = 64;            // K-slice per LDS stage (2 WMMA k-steps)

typedef __attribute__((ext_vector_type(16))) __bf16        bf16x16;
typedef __attribute__((ext_vector_type(8)))  __bf16        bf16x8;
typedef __attribute__((ext_vector_type(8)))  float         floatx8;
typedef __attribute__((ext_vector_type(4)))  float         floatx4;
typedef __attribute__((ext_vector_type(4)))  unsigned int  uintx4;
typedef __attribute__((ext_vector_type(8)))  int           intx8;
typedef __attribute__((ext_vector_type(4)))  int           intx4;
typedef __attribute__((ext_vector_type(4)))  unsigned short ushortx4;

union FragAB { bf16x16 v; bf16x8 h[2]; };

#if __has_builtin(__builtin_amdgcn_tensor_load_to_lds)
#define HAVE_TDM 1
#else
#define HAVE_TDM 0
#endif

DEVINL unsigned short f2bf_bits(float f) {
  unsigned u = __float_as_uint(f);
  unsigned r = u + 0x7FFFu + ((u >> 16) & 1u);   // round-to-nearest-even
  return (unsigned short)(r >> 16);
}

// --------- CDNA5 async memory->LDS copy (ASYNCcnt-tracked), 64B per call
// ISA §15.18: each b128 copies 16 bytes from global memory into LDS at the same offset
DEVINL void async_cp64(unsigned lds_off, const void* gsrc) {
  unsigned long long ga = (unsigned long long)gsrc;
  asm volatile("global_load_async_to_lds_b128 %0, %1, off"
               :: "v"(lds_off), "v"(ga) : "memory");
  asm volatile("global_load_async_to_lds_b128 %0, %1, off offset:16"
               :: "v"(lds_off), "v"(ga) : "memory");
  asm volatile("global_load_async_to_lds_b128 %0, %1, off offset:32"
               :: "v"(lds_off), "v"(ga) : "memory");
  asm volatile("global_load_async_to_lds_b128 %0, %1, off offset:48"
               :: "v"(lds_off), "v"(ga) : "memory");
}

DEVINL void wait_async0() {
#if __has_builtin(__builtin_amdgcn_s_wait_asynccnt)
  __builtin_amdgcn_s_wait_asynccnt(0);
#else
  asm volatile("s_wait_asynccnt 0x0" ::: "memory");
#endif
}

DEVINL void wait_tensor0() {
#if __has_builtin(__builtin_amdgcn_s_wait_tensorcnt)
  __builtin_amdgcn_s_wait_tensorcnt(0);
#else
  asm volatile("s_wait_tensorcnt 0x0" ::: "memory");
#endif
}

DEVINL unsigned lds_off_of(const void* p) {
  // Flat shared addresses carry the LDS byte offset in their low 32 bits.
  return (unsigned)(size_t)p;
}

#if HAVE_TDM
// TDM: load a 2D tile [tile_dim1=128 rows x tile_dim0=64 elems] of 2-byte data,
// row stride strideElems, packed row-major into LDS at lds_addr. D# per ISA §8.
DEVINL void tdm_load_tile_128x64(unsigned lds_addr, const void* g, unsigned strideElems) {
  unsigned long long ga = (unsigned long long)g;
  uintx4 g0;
  g0.x = 1u;                                     // count=1 (valid), user mode
  g0.y = lds_addr;                               // lds_addr [63:32]
  g0.z = (unsigned)ga;                           // global_addr low
  g0.w = (unsigned)(ga >> 32) | (2u << 30);      // global_addr hi | type=2
  intx8 g1;
  g1[0] = 0x00010000;                            // wg_mask=0, data_size=1 (2B)
  g1[1] = (int)0xFFFF0000;                       // tensor_dim0[15:0]=0xFFFF
  g1[2] = (int)0xFFFF7FFF;                       // dim0_hi=0x7FFF, dim1_lo=0xFFFF
  g1[3] = (int)(0x00007FFF | (BK << 16));        // dim1_hi, tile_dim0=64
  g1[4] = 128;                                   // tile_dim1=128, tile_dim2=0
  g1[5] = (int)strideElems;                      // tensor_dim0_stride low
  g1[6] = 0;
  g1[7] = 0;
  intx4 gz4; gz4.x = 0; gz4.y = 0; gz4.z = 0; gz4.w = 0;
  intx8 gz8;
#pragma unroll
  for (int i = 0; i < 8; i++) gz8[i] = 0;
  // 6-arg form (clang-23 / therock headers): (g0, g1, g2, g3, g4, cpol)
  __builtin_amdgcn_tensor_load_to_lds(g0, g1, gz4, gz4, gz8, 0);
}
#endif

// ---------------------------------------------------------------- gating
__global__ __launch_bounds__(256) void gate_topk_kernel(
    const float* __restrict__ x, const float* __restrict__ gw,
    const float* __restrict__ gb, int* __restrict__ topk_i,
    float* __restrict__ topk_p)
{
  int gtid = blockIdx.x * blockDim.x + threadIdx.x;
  int tok  = gtid >> 5;
  int lane = threadIdx.x & 31;
  if (tok >= Btok) return;

  const float* xr = x + (size_t)tok * Dim;
  float acc[Eexp];
#pragma unroll
  for (int e = 0; e < Eexp; e++) acc[e] = 0.0f;

  for (int d = lane; d < Dim; d += 32) {
    float xv = xr[d];
    const float* wr = gw + (size_t)d * Eexp;
#pragma unroll
    for (int e = 0; e < Eexp; e++) acc[e] += xv * wr[e];
  }
#pragma unroll
  for (int e = 0; e < Eexp; e++) {
#pragma unroll
    for (int off = 16; off; off >>= 1)
      acc[e] += __shfl_xor(acc[e], off, 32);
  }

  if (lane == 0) {
    float lg[Eexp], p[Eexp];
    float mx = -1e30f;
#pragma unroll
    for (int e = 0; e < Eexp; e++) { lg[e] = acc[e] + gb[e]; mx = fmaxf(mx, lg[e]); }
    float s = 0.0f;
#pragma unroll
    for (int e = 0; e < Eexp; e++) { p[e] = __expf(lg[e] - mx); s += p[e]; }
    float inv = 1.0f / s;

    int i0 = 0; float p0 = p[0];
#pragma unroll
    for (int e = 1; e < Eexp; e++) if (p[e] > p0) { p0 = p[e]; i0 = e; }
    int i1 = -1; float p1 = -1.0f;
#pragma unroll
    for (int e = 0; e < Eexp; e++) if (e != i0 && p[e] > p1) { p1 = p[e]; i1 = e; }

    topk_i[tok * 2 + 0] = i0;
    topk_i[tok * 2 + 1] = i1;
    topk_p[tok * 2 + 0] = p0 * inv;
    topk_p[tok * 2 + 1] = p1 * inv;
  }
}

// --------------------------------------------------- routing: positions
__global__ __launch_bounds__(1024) void hist_kernel(
    const int* __restrict__ topk_i, int* __restrict__ blockCounts)
{
  __shared__ int cnt[Eexp];
  int tid = threadIdx.x;
  if (tid < Eexp) cnt[tid] = 0;
  __syncthreads();
  int n = blockIdx.x * 1024 + tid;
  atomicAdd(&cnt[topk_i[n]], 1);
  __syncthreads();
  if (tid < Eexp) blockCounts[blockIdx.x * Eexp + tid] = cnt[tid];
}

__global__ void scan_kernel(const int* __restrict__ bc, int* __restrict__ bo)
{
  int e = threadIdx.x;
  if (e >= Eexp) return;
  int run = 0;
  for (int b = 0; b < NBLK; b++) {
    bo[b * Eexp + e] = run;
    run += bc[b * Eexp + e];
  }
}

__global__ void init_slots_kernel(int* __restrict__ slot_token)
{
  int i = blockIdx.x * blockDim.x + threadIdx.x;
  if (i < Eexp * Cap) slot_token[i] = -1;
}

__global__ void zero_out_kernel(float* __restrict__ out)
{
  size_t i = (size_t)(blockIdx.x * blockDim.x + threadIdx.x) * 4;
  if (i < (size_t)Btok * Odim) {
    floatx4 z; z.x = 0.f; z.y = 0.f; z.z = 0.f; z.w = 0.f;
    *(floatx4*)(out + i) = z;
  }
}

__global__ __launch_bounds__(1024) void assign_kernel(
    const int* __restrict__ topk_i, const float* __restrict__ topk_p,
    const int* __restrict__ bo, int* __restrict__ slot_token,
    float* __restrict__ slot_w)
{
  __shared__ unsigned char se[1024];
  __shared__ int base[Eexp];
  int tid = threadIdx.x;
  int n = blockIdx.x * 1024 + tid;
  int e = topk_i[n];
  se[tid] = (unsigned char)e;
  if (tid < Eexp) base[tid] = bo[blockIdx.x * Eexp + tid];
  __syncthreads();
  int rank = 0;
  for (int j = 0; j < tid; j++) rank += (se[j] == (unsigned char)e) ? 1 : 0;
  int pos = base[e] + rank;                    // flat-order position in expert queue
  if (pos < Cap) {
    slot_token[e * Cap + pos] = n >> 1;        // sample index (n / top_k)
    slot_w[e * Cap + pos]     = topk_p[n];
  }
}

// ------------------------------------------------- dispatch (fp32->bf16)
__global__ __launch_bounds__(256) void dispatch_kernel(
    const float* __restrict__ x, const int* __restrict__ slot_token,
    unsigned short* __restrict__ bufA)
{
  int slot = blockIdx.x;                 // 0 .. E*Cap-1
  int t = slot_token[slot];
  int c0 = threadIdx.x * 4;
  ushortx4 o;
  if (t >= 0) {
    floatx4 v = *(const floatx4*)(x + (size_t)t * Dim + c0);
    o.x = f2bf_bits(v.x); o.y = f2bf_bits(v.y);
    o.z = f2bf_bits(v.z); o.w = f2bf_bits(v.w);
  } else {
    o.x = 0; o.y = 0; o.z = 0; o.w = 0;
  }
  *(ushortx4*)(bufA + (size_t)slot * Dim + c0) = o;
}

// -------------------------------- weight convert + transpose fp32->bf16
// in: [E, Rr, Cc] f32 row-major  ->  out: [E, Cc, Rr] bf16 row-major
__global__ __launch_bounds__(256) void transposeconv_kernel(
    const float* __restrict__ in, unsigned short* __restrict__ out,
    int Rr, int Cc)
{
  __shared__ float tile[32][33];
  int e  = blockIdx.z;
  int c0 = blockIdx.x * 32;
  int r0 = blockIdx.y * 32;
  const float* ine = in + (size_t)e * Rr * Cc;
  unsigned short* oute = out + (size_t)e * Rr * Cc;
  int tx = threadIdx.x & 31, ty = threadIdx.x >> 5;
#pragma unroll
  for (int i = 0; i < 4; i++) {
    int r = r0 + ty + i * 8;
    tile[ty + i * 8][tx] = ine[(size_t)r * Cc + c0 + tx];
  }
  __syncthreads();
#pragma unroll
  for (int i = 0; i < 4; i++) {
    int c = c0 + ty + i * 8;
    oute[(size_t)c * Rr + r0 + tx] = f2bf_bits(tile[tx][ty + i * 8]);
  }
}

// ------------------------------------------------------- grouped GEMMs
// A:  [E, Cap, Kd] bf16 row-major      (staged via per-lane async-to-LDS)
// Bt: [E, N,  Kd] bf16 row-major      (staged via TDM tensor_load_to_lds)
// mode 0: D = GELU(A*B + b)   -> outBf [E, Cap, N] bf16
// mode 1: D = A*B + b, then scatter: out[token] += slot_w * D  (atomicAdd)
__global__ __launch_bounds__(256) void moe_gemm_kernel(
    const unsigned short* __restrict__ A,
    const unsigned short* __restrict__ Bt,
    const float* __restrict__ bias,
    unsigned short* __restrict__ outBf,
    float* __restrict__ outF,
    const int* __restrict__ slot_token,
    const float* __restrict__ slot_w,
    int Kd, int N, int mode)
{
  __shared__ unsigned short sA[2][128 * BK];   // 2 x 16KB
  __shared__ unsigned short sB[2][128 * BK];   // 2 x 16KB

  const int e    = blockIdx.z;
  const int bm   = blockIdx.x;
  const int bn   = blockIdx.y;
  const int tid  = threadIdx.x;
  const int lane = tid & 31;
  const int wv   = tid >> 5;         // 8 waves: 2 (M) x 4 (N)
  const int wm0  = (wv >> 2) * 64;
  const int wn0  = (wv & 3) * 32;
  const int half = lane >> 4;
  const int lr   = lane & 15;

  const unsigned short* Ae  = A  + (size_t)e * Cap * Kd + (size_t)bm * 128 * Kd;
  const unsigned short* Bte = Bt + (size_t)e * N   * Kd + (size_t)bn * 128 * Kd;

  floatx8 acc[4][2];
#pragma unroll
  for (int i = 0; i < 4; i++)
#pragma unroll
    for (int j = 0; j < 2; j++)
#pragma unroll
      for (int r = 0; r < 8; r++) acc[i][j][r] = 0.0f;

  // staging map: 128 rows x BK(64) cols; thread -> (row, 32-element half-row)
  const int srow = tid >> 1;          // 0..127
  const int soff = (tid & 1) * 32;    // element offset inside BK slice

  const unsigned sAoff0 = lds_off_of(&sA[0][srow * BK + soff]);
  const unsigned bufStride = 128 * BK * 2;   // bytes between double buffers
#if HAVE_TDM
  const unsigned sBbase = lds_off_of(&sB[0][0]);
#else
  const unsigned sBoff0 = lds_off_of(&sB[0][srow * BK + soff]);
#endif

  // prologue: stage 0 into buffer 0
  async_cp64(sAoff0, Ae + (size_t)srow * Kd + soff);
#if HAVE_TDM
  if (wv == 0) tdm_load_tile_128x64(sBbase, Bte, (unsigned)Kd);
#else
  async_cp64(sBoff0, Bte + (size_t)srow * Kd + soff);
#endif

  int buf = 0;
  for (int k0 = 0; k0 < Kd; k0 += BK, buf ^= 1) {
    wait_async0();        // this wave's async writes for stage k0 are done
#if HAVE_TDM
    if (wv == 0) wait_tensor0();   // wave 0's TDM transfer for stage k0 is done
#endif
    __syncthreads();      // publish stage k0 to all waves
    if (k0 + BK < Kd) {   // stage k0+BK into the other buffer
      async_cp64(sAoff0 + (buf ^ 1) * bufStride,
                 Ae + (size_t)srow * Kd + k0 + BK + soff);
#if HAVE_TDM
      if (wv == 0)
        tdm_load_tile_128x64(sBbase + (buf ^ 1) * bufStride,
                             Bte + (size_t)(k0 + BK), (unsigned)Kd);
#else
      async_cp64(sBoff0 + (buf ^ 1) * bufStride,
                 Bte + (size_t)srow * Kd + k0 + BK + soff);
#endif
    }

#pragma unroll
    for (int kk = 0; kk < BK; kk += 32) {
      // B fragments: lane<16 holds K 0..15 of column n; lane>=16 holds K 16..31
      FragAB bfr[2];
#pragma unroll
      for (int ni = 0; ni < 2; ni++) {
        int n = wn0 + ni * 16 + lr;
        bfr[ni].h[0] = *(const bf16x8*)&sB[buf][n * BK + kk + half * 16];
        bfr[ni].h[1] = *(const bf16x8*)&sB[buf][n * BK + kk + half * 16 + 8];
      }
      // A fragments: lane<16 holds K 0..15 of row m; lane>=16 holds K 16..31
#pragma unroll
      for (int mi = 0; mi < 4; mi++) {
        FragAB afr;
        int m = wm0 + mi * 16 + lr;
        afr.h[0] = *(const bf16x8*)&sA[buf][m * BK + kk + half * 16];
        afr.h[1] = *(const bf16x8*)&sA[buf][m * BK + kk + half * 16 + 8];
#pragma unroll
        for (int ni = 0; ni < 2; ni++) {
          acc[mi][ni] = __builtin_amdgcn_wmma_f32_16x16x32_bf16(
              false, afr.v, false, bfr[ni].v, (short)0, acc[mi][ni], false, false);
        }
      }
    }
  }

  // epilogue: C/D layout — VGPR r, lanes 0..15 => M=r, lanes 16..31 => M=r+8; N=lane&15
#pragma unroll
  for (int mi = 0; mi < 4; mi++) {
#pragma unroll
    for (int ni = 0; ni < 2; ni++) {
#pragma unroll
      for (int r = 0; r < 8; r++) {
        int ml = wm0 + mi * 16 + r + half * 8;
        int nl = wn0 + ni * 16 + lr;
        int gm = bm * 128 + ml;
        int gn = bn * 128 + nl;
        float v = acc[mi][ni][r] + bias[e * N + gn];
        if (mode == 0) {
          v = 0.5f * v * (1.0f + erff(v * 0.70710678118654752f));   // exact GELU
          outBf[(size_t)e * Cap * N + (size_t)gm * N + gn] = f2bf_bits(v);
        } else {
          int t = slot_token[e * Cap + gm];
          if (t >= 0) {
            atomicAdd(&outF[(size_t)t * N + gn], v * slot_w[e * Cap + gm]);
          }
        }
      }
    }
  }
}

// ---------------------------------------------------------------- launch
extern "C" void kernel_launch(void* const* d_in, const int* in_sizes, int n_in,
                              void* d_out, int out_size, void* d_ws, size_t ws_size,
                              hipStream_t stream)
{
  const float* x  = (const float*)d_in[0];
  const float* gw = (const float*)d_in[1];
  const float* gb = (const float*)d_in[2];
  const float* w1 = (const float*)d_in[3];
  const float* b1 = (const float*)d_in[4];
  const float* w2 = (const float*)d_in[5];
  const float* b2 = (const float*)d_in[6];
  float* out = (float*)d_out;

  char* wsp = (char*)d_ws;
  auto alloc = [&](size_t bytes) -> char* {
    char* p = wsp;
    wsp += (bytes + 255) & ~(size_t)255;
    return p;
  };

  int*   topk_i      = (int*)  alloc(sizeof(int)   * Nrt);
  float* topk_p      = (float*)alloc(sizeof(float) * Nrt);
  int*   blockCounts = (int*)  alloc(sizeof(int)   * NBLK * Eexp);
  int*   blockOffs   = (int*)  alloc(sizeof(int)   * NBLK * Eexp);
  int*   slot_token  = (int*)  alloc(sizeof(int)   * Eexp * Cap);
  float* slot_w      = (float*)alloc(sizeof(float) * Eexp * Cap);
  unsigned short* bufA = (unsigned short*)alloc(2ull * Eexp * Cap  * Dim);
  unsigned short* w1T  = (unsigned short*)alloc(2ull * Eexp * Dim  * Hdim);
  unsigned short* w2T  = (unsigned short*)alloc(2ull * Eexp * Hdim * Odim);
  unsigned short* Hbuf = (unsigned short*)alloc(2ull * Eexp * Cap  * Hdim);

  zero_out_kernel<<<(Btok * Odim / 4 + 255) / 256, 256, 0, stream>>>(out);
  init_slots_kernel<<<(Eexp * Cap + 255) / 256, 256, 0, stream>>>(slot_token);
  gate_topk_kernel<<<(Btok * 32) / 256, 256, 0, stream>>>(x, gw, gb, topk_i, topk_p);
  hist_kernel<<<NBLK, 1024, 0, stream>>>(topk_i, blockCounts);
  scan_kernel<<<1, 32, 0, stream>>>(blockCounts, blockOffs);
  assign_kernel<<<NBLK, 1024, 0, stream>>>(topk_i, topk_p, blockOffs, slot_token, slot_w);
  dispatch_kernel<<<Eexp * Cap, 256, 0, stream>>>(x, slot_token, bufA);
  transposeconv_kernel<<<dim3(Hdim / 32, Dim / 32, Eexp), 256, 0, stream>>>(w1, w1T, Dim, Hdim);
  transposeconv_kernel<<<dim3(Odim / 32, Hdim / 32, Eexp), 256, 0, stream>>>(w2, w2T, Hdim, Odim);

  moe_gemm_kernel<<<dim3(Cap / 128, Hdim / 128, Eexp), 256, 0, stream>>>(
      bufA, w1T, b1, Hbuf, nullptr, nullptr, nullptr, Dim, Hdim, 0);
  moe_gemm_kernel<<<dim3(Cap / 128, Odim / 128, Eexp), 256, 0, stream>>>(
      Hbuf, w2T, b2, nullptr, out, slot_token, slot_w, Hdim, Odim, 1);
}
